// DisentangledSelfAttention_55319178773197
// MI455X (gfx1250) — compile-verified, matrix operand order checked by reading, not run on hardware
//
#include <hip/hip_runtime.h>
#include <math.h>

// ---------------------------------------------------------------------------
// DeBERTa disentangled self-attention for MI455X (gfx1250, wave32, WMMA)
// B=4 S=1024 H=1024 NH=16 HD=64, 512 rel-pos buckets.
// All GEMMs use v_wmma_f32_16x16x32_f16 (f16 in, f32 accumulate).
// V tiles are staged to LDS with gfx1250 async copies (ASYNCcnt) and the
// next K tile is warmed with global_prefetch_b8.
// ---------------------------------------------------------------------------

typedef __attribute__((ext_vector_type(16))) _Float16 v16h;
typedef __attribute__((ext_vector_type(8)))  _Float16 v8h;
typedef __attribute__((ext_vector_type(8)))  float    v8f;

#define WMMA_F16(a, b, c) \
  __builtin_amdgcn_wmma_f32_16x16x32_f16(false, (a), false, (b), (short)0, (c), false, false)

// Build a v16h A/B fragment from two contiguous 8-half (16B) chunks.
// ISA 16-bit fragment layout: lane holds K = {8*hi+0..7, 16+8*hi+0..7}.
__device__ __forceinline__ v16h frag_ld2(const _Float16* p0, const _Float16* p1) {
  union { v16h v; v8h h[2]; } u;
  u.h[0] = *(const v8h*)p0;
  u.h[1] = *(const v8h*)p1;
  return u.v;
}

// gfx1250 async global->LDS 16-byte copy (per active lane), tracked by ASYNCcnt.
// lds_byte_addr: LDS byte address (low 32 bits of flat shared pointer),
// gaddr: 64-bit global address.
__device__ __forceinline__ void async_g2l_b128(unsigned lds_byte_addr, const void* gaddr) {
  asm volatile("global_load_async_to_lds_b128 %0, %1, off"
               :: "v"(lds_byte_addr), "v"(gaddr)
               : "memory");
}

__device__ __forceinline__ void wait_asynccnt0() {
  asm volatile("s_wait_asynccnt 0" ::: "memory");
}

// ---------------------------------------------------------------------------
// Kernel: delta -> clipped log-bucket index table (2047 entries, f64 to match
// the numpy reference exactly).  idx(q-k) serves both c2p and p2c gathers.
// ---------------------------------------------------------------------------
__global__ void build_idx_kernel(int* __restrict__ idx) {
  int i = blockIdx.x * blockDim.x + threadIdx.x;
  if (i >= 2047) return;
  const int delta = i - 1023;                 // q - k
  const int mid = 128;                        // BUCKETS/2
  double ad = (double)(delta < 0 ? -delta : delta);
  double abs_pos = (delta < mid && delta > -mid) ? (double)(mid - 1) : ad;
  double log_pos = ceil(log(abs_pos / mid) / log((1024.0 - 1.0) / mid) * (mid - 1)) + mid;
  double sign = (delta > 0) ? 1.0 : ((delta < 0) ? -1.0 : 0.0);
  double bucket = (abs_pos <= (double)mid) ? (double)delta : log_pos * sign;
  int bi = (int)bucket + 256;                 // + ATT_SPAN
  bi = bi < 0 ? 0 : (bi > 511 ? 511 : bi);
  idx[i] = bi;
}

// ---------------------------------------------------------------------------
// Kernel: f32 -> f16 conversion
// ---------------------------------------------------------------------------
__global__ void f32_to_f16_kernel(const float* __restrict__ src,
                                  _Float16* __restrict__ dst, int n) {
  int i = blockIdx.x * blockDim.x + threadIdx.x;
  if (i < n) dst[i] = (_Float16)src[i];
}

// ---------------------------------------------------------------------------
// Kernel: Y[m,n] = (sum_k A[m,k] * W[n,k] + bias[n]) * scale   (f16 out)
// K = N = 1024. Block = 128 threads (4 waves); each wave computes a 16x64
// strip of a 64x64 output tile via WMMA.
// ---------------------------------------------------------------------------
__global__ void __launch_bounds__(128)
proj_gemm_kernel(const _Float16* __restrict__ A, const _Float16* __restrict__ W,
                 const float* __restrict__ bias, _Float16* __restrict__ out,
                 float scale) {
  const int K = 1024;
  const int w = threadIdx.x >> 5, l = threadIdx.x & 31;
  const int hi = l >> 4, n16 = l & 15;
  const int m0 = blockIdx.x * 64 + w * 16;
  const int n0 = blockIdx.y * 64;

  v8f acc[4] = {};
  const _Float16* arow = A + (size_t)(m0 + n16) * K;
  for (int kk = 0; kk < K; kk += 32) {
    v16h af = frag_ld2(arow + kk + hi * 8, arow + kk + 16 + hi * 8);
#pragma unroll
    for (int t = 0; t < 4; ++t) {
      const _Float16* wrow = W + (size_t)(n0 + t * 16 + n16) * K + kk;
      v16h bf = frag_ld2(wrow + hi * 8, wrow + 16 + hi * 8);
      acc[t] = WMMA_F16(af, bf, acc[t]);
    }
  }
#pragma unroll
  for (int t = 0; t < 4; ++t) {
    int n = n0 + t * 16 + n16;
    float bv = bias[n];
#pragma unroll
    for (int v = 0; v < 8; ++v) {
      int m = m0 + v + 8 * hi;
      out[(size_t)m * K + n] = (_Float16)((acc[t][v] + bv) * scale);
    }
  }
}

// ---------------------------------------------------------------------------
// Kernel: position scores.  out[bh, m, p] = sum_d QK[b,m,h,d] * POS[p,h,d]
// (M=1024 per (b,h), N=512, K=64).  Output f16.
// ---------------------------------------------------------------------------
__global__ void __launch_bounds__(128)
posscore_gemm_kernel(const _Float16* __restrict__ QK,   // [B*S, H]
                     const _Float16* __restrict__ POS,  // [512, H]
                     _Float16* __restrict__ out) {      // [B*NH, 1024, 512]
  const int w = threadIdx.x >> 5, l = threadIdx.x & 31;
  const int hi = l >> 4, n16 = l & 15;
  const int bh = blockIdx.z, b = bh >> 4, h = bh & 15;
  const int m0 = blockIdx.x * 64 + w * 16;
  const int p0 = blockIdx.y * 64;

  v8f acc[4] = {};
  const _Float16* arow = QK + ((size_t)(b * 1024 + m0 + n16)) * 1024 + h * 64;
#pragma unroll
  for (int kk = 0; kk < 64; kk += 32) {
    v16h af = frag_ld2(arow + kk + hi * 8, arow + kk + 16 + hi * 8);
#pragma unroll
    for (int t = 0; t < 4; ++t) {
      const _Float16* prow = POS + (size_t)(p0 + t * 16 + n16) * 1024 + h * 64 + kk;
      v16h bf = frag_ld2(prow + hi * 8, prow + 16 + hi * 8);
      acc[t] = WMMA_F16(af, bf, acc[t]);
    }
  }
#pragma unroll
  for (int t = 0; t < 4; ++t)
#pragma unroll
    for (int v = 0; v < 8; ++v) {
      int m = m0 + v + 8 * hi;
      out[((size_t)bh * 1024 + m) * 512 + p0 + t * 16 + n16] = (_Float16)acc[t][v];
    }
}

// ---------------------------------------------------------------------------
// Kernel: fused attention.  One workgroup = (b, h, 64-row q tile); 4 waves,
// each owns 16 q rows.  Flash-style streaming softmax over 16 k-tiles of 64.
// Q is pre-scaled by 1/sqrt(192); pos_q likewise, so scores need no scaling.
// V tiles stream through LDS via gfx1250 async global->LDS copies.
// ---------------------------------------------------------------------------
__global__ void __launch_bounds__(128)
attn_kernel(const _Float16* __restrict__ Q,    // [B*S, H] (pre-scaled)
            const _Float16* __restrict__ K,    // [B*S, H]
            const _Float16* __restrict__ V,    // [B*S, H]
            const _Float16* __restrict__ c2p,  // [B*NH, S, 512] (pre-scaled)
            const _Float16* __restrict__ p2c,  // [B*NH, S, 512] (pre-scaled)
            const int* __restrict__ amask,     // [B,1,S,S]
            const int* __restrict__ idxtab,    // [2047]
            float* __restrict__ out) {         // [B,S,H]
  __shared__ _Float16 sv[64 * 64];          // V tile (shared by all waves)
  __shared__ float    ss[4][16 * 64];       // per-wave score strip
  __shared__ _Float16 sp[4][16 * 64];       // per-wave prob strip (f16)
  __shared__ short    sidx[2048];           // delta -> bucket index
  __shared__ float    srm[4][16], srl[4][16], srs[4][16];  // row max/sum/scale

  const int tid = threadIdx.x;
  const int w = tid >> 5, l = tid & 31;
  const int hi = l >> 4, n16 = l & 15;
  const int q0 = blockIdx.x * 64;
  const int h = blockIdx.y;
  const int b = blockIdx.z;

  for (int i = tid; i < 2047; i += 128) sidx[i] = (short)idxtab[i];
  if (l < 16) { srm[w][l] = -INFINITY; srl[w][l] = 0.0f; }

  // Q fragments for this wave's 16 rows (d split 0..31 / 32..63), kept in regs
  const _Float16* qrow = Q + ((size_t)(b * 1024 + q0 + w * 16 + n16)) * 1024 + h * 64;
  v16h qa0 = frag_ld2(qrow + hi * 8, qrow + 16 + hi * 8);
  v16h qa1 = frag_ld2(qrow + 32 + hi * 8, qrow + 48 + hi * 8);

  // LDS byte address of the V tile (flat shared pointer low 32 bits == LDS offset)
  const unsigned sv_base = (unsigned)(size_t)(void*)sv;

  v8f o[4] = {};
  __syncthreads();

  const size_t bh = (size_t)(b * 16 + h);
  const _Float16* c2pq = c2p + (bh * 1024 + q0 + w * 16) * 512;
  const int* mrow = amask + ((size_t)b * 1024 + q0 + w * 16) * 1024;

  for (int kt = 0; kt < 16; ++kt) {
    const int k0 = kt * 64;

    // ---- async-stage V tile (64 rows x 128B) to LDS: 512 x 16B chunks ----
#pragma unroll
    for (int j = 0; j < 4; ++j) {
      int i = tid + 128 * j;               // chunk id 0..511
      int r = i >> 3, cc = i & 7;          // row, 16B-chunk within row
      const void* g = (const void*)(V + ((size_t)(b * 1024 + k0 + r)) * 1024 + h * 64 + cc * 8);
      async_g2l_b128(sv_base + (unsigned)(r * 128 + cc * 16), g);
    }

    // ---- prefetch next K tile (one 128B row segment per thread) ----
    if (kt < 15 && tid < 64) {
      __builtin_prefetch((const void*)(K + ((size_t)(b * 1024 + k0 + 64 + tid)) * 1024 + h * 64), 0, 3);
    }

    // ---- scores: 16x64 strip = Q(16x64) @ K(64..)^T via WMMA ----
    const _Float16* kbase = K + ((size_t)(b * 1024 + k0)) * 1024 + h * 64;
#pragma unroll
    for (int t = 0; t < 4; ++t) {
      const _Float16* krow = kbase + (size_t)(t * 16 + n16) * 1024;
      v16h kb0 = frag_ld2(krow + hi * 8, krow + 16 + hi * 8);
      v16h kb1 = frag_ld2(krow + 32 + hi * 8, krow + 48 + hi * 8);
      v8f c = {};
      c = WMMA_F16(qa0, kb0, c);
      c = WMMA_F16(qa1, kb1, c);
#pragma unroll
      for (int v = 0; v < 8; ++v)
        ss[w][(v + 8 * hi) * 64 + t * 16 + n16] = c[v];
    }

    // ---- add gathered c2p + p2c terms, apply mask ----
    for (int e = l; e < 1024; e += 32) {
      int r = e >> 6, c = e & 63;
      int qq = q0 + w * 16 + r;
      int kc = k0 + c;
      float s = ss[w][e];
      int p = sidx[qq - kc + 1023];
      s += (float)c2pq[(size_t)r * 512 + p];
      s += (float)p2c[(bh * 1024 + kc) * 512 + p];
      if (mrow[(size_t)r * 1024 + kc] == 0) s = -3.0e38f;
      ss[w][e] = s;
    }

    // ---- streaming softmax row update (2 lanes per row) ----
    {
      int r = l >> 1, hh = l & 1;
      const float* row = &ss[w][r * 64 + hh * 32];
      float tmax = -INFINITY;
#pragma unroll
      for (int c = 0; c < 32; ++c) tmax = fmaxf(tmax, row[c]);
      tmax = fmaxf(tmax, __shfl_xor(tmax, 1, 32));
      float mold = srm[w][r];
      float mnew = fmaxf(mold, tmax);
      float psum = 0.0f;
      _Float16* prow = &sp[w][r * 64 + hh * 32];
#pragma unroll
      for (int c = 0; c < 32; ++c) {
        float pv = __expf(row[c] - mnew);
        prow[c] = (_Float16)pv;
        psum += pv;
      }
      psum += __shfl_xor(psum, 1, 32);
      if (hh == 0) {
        float sc = __expf(mold - mnew);
        srl[w][r] = srl[w][r] * sc + psum;
        srm[w][r] = mnew;
        srs[w][r] = sc;
      }
    }

    // ---- rescale accumulator by per-row exp(mold-mnew) ----
#pragma unroll
    for (int v = 0; v < 8; ++v) {
      float sc = srs[w][v + 8 * hi];
#pragma unroll
      for (int t = 0; t < 4; ++t) o[t][v] *= sc;
    }

    // ---- make sure async V tile landed, then sync all waves ----
    wait_asynccnt0();
    __syncthreads();

    // ---- O += P(16x64) @ V(64x64) via WMMA ----
    {
      const _Float16* pr = &sp[w][n16 * 64];
      v16h pa0 = frag_ld2(pr + hi * 8, pr + 16 + hi * 8);
      v16h pa1 = frag_ld2(pr + 32 + hi * 8, pr + 48 + hi * 8);
#pragma unroll
      for (int t = 0; t < 4; ++t) {
        int d = t * 16 + n16;
        v16h vb0, vb1;
#pragma unroll
        for (int j = 0; j < 8; ++j) {
          vb0[j]     = sv[(8 * hi + j) * 64 + d];
          vb0[j + 8] = sv[(16 + 8 * hi + j) * 64 + d];
          vb1[j]     = sv[(32 + 8 * hi + j) * 64 + d];
          vb1[j + 8] = sv[(48 + 8 * hi + j) * 64 + d];
        }
        o[t] = WMMA_F16(pa0, vb0, o[t]);
        o[t] = WMMA_F16(pa1, vb1, o[t]);
      }
    }
    __syncthreads();   // protect sv/sp before next iteration overwrites
  }

  // ---- epilogue: out[b, q, h*64+d] = O / rowsum ----
#pragma unroll
  for (int v = 0; v < 8; ++v) {
    int r = v + 8 * hi;
    float inv = 1.0f / srl[w][r];
    int qq = q0 + w * 16 + r;
    float* orow = out + ((size_t)(b * 1024 + qq)) * 1024 + h * 64;
#pragma unroll
    for (int t = 0; t < 4; ++t)
      orow[t * 16 + n16] = o[t][v] * inv;
  }
}

// ---------------------------------------------------------------------------
// Host launcher
// ---------------------------------------------------------------------------
extern "C" void kernel_launch(void* const* d_in, const int* in_sizes, int n_in,
                              void* d_out, int out_size, void* d_ws, size_t ws_size,
                              hipStream_t stream) {
  (void)in_sizes; (void)n_in; (void)out_size; (void)ws_size;

  const float* hs  = (const float*)d_in[0];   // [4,1024,1024]
  const int*   am  = (const int*)d_in[1];     // [4,1,1024,1024]
  const float* rel = (const float*)d_in[2];   // [512,1024]
  const float* Wq  = (const float*)d_in[3];
  const float* bq  = (const float*)d_in[4];
  const float* Wk  = (const float*)d_in[5];
  const float* bk  = (const float*)d_in[6];
  const float* Wv  = (const float*)d_in[7];
  const float* bv  = (const float*)d_in[8];
  float* out = (float*)d_out;

  char* ws = (char*)d_ws;
  size_t off = 0;
  auto wsalloc = [&](size_t bytes) -> void* {
    void* p = ws + off;
    off = (off + bytes + 255) & ~(size_t)255;
    return p;
  };

  int*      idxt = (int*)     wsalloc(2048 * sizeof(int));
  _Float16* Xh   = (_Float16*)wsalloc((size_t)4096 * 1024 * 2);
  _Float16* Wqh  = (_Float16*)wsalloc((size_t)1024 * 1024 * 2);
  _Float16* Wkh  = (_Float16*)wsalloc((size_t)1024 * 1024 * 2);
  _Float16* Wvh  = (_Float16*)wsalloc((size_t)1024 * 1024 * 2);
  _Float16* posh = (_Float16*)wsalloc((size_t)512 * 1024 * 2);
  _Float16* Qh   = (_Float16*)wsalloc((size_t)4096 * 1024 * 2);
  _Float16* Kh   = (_Float16*)wsalloc((size_t)4096 * 1024 * 2);
  _Float16* Vh   = (_Float16*)wsalloc((size_t)4096 * 1024 * 2);
  _Float16* pkh  = (_Float16*)wsalloc((size_t)512 * 1024 * 2);
  _Float16* pqh  = (_Float16*)wsalloc((size_t)512 * 1024 * 2);
  _Float16* c2pw = (_Float16*)wsalloc((size_t)64 * 1024 * 512 * 2);
  _Float16* p2cw = (_Float16*)wsalloc((size_t)64 * 1024 * 512 * 2);

  // 1) bucket-index table
  build_idx_kernel<<<dim3(8), dim3(256), 0, stream>>>(idxt);

  // 2) f32 -> f16 conversions
  f32_to_f16_kernel<<<(4194304 + 255) / 256, 256, 0, stream>>>(hs, Xh, 4194304);
  f32_to_f16_kernel<<<(1048576 + 255) / 256, 256, 0, stream>>>(Wq, Wqh, 1048576);
  f32_to_f16_kernel<<<(1048576 + 255) / 256, 256, 0, stream>>>(Wk, Wkh, 1048576);
  f32_to_f16_kernel<<<(1048576 + 255) / 256, 256, 0, stream>>>(Wv, Wvh, 1048576);
  f32_to_f16_kernel<<<(524288 + 255) / 256, 256, 0, stream>>>(rel, posh, 524288);

  // 3) projections (scale 1/sqrt(HD*3) folded into Q and pos_q)
  const float iscale = 1.0f / sqrtf(64.0f * 3.0f);
  proj_gemm_kernel<<<dim3(64, 16), 128, 0, stream>>>(Xh, Wqh, bq, Qh, iscale);
  proj_gemm_kernel<<<dim3(64, 16), 128, 0, stream>>>(Xh, Wkh, bk, Kh, 1.0f);
  proj_gemm_kernel<<<dim3(64, 16), 128, 0, stream>>>(Xh, Wvh, bv, Vh, 1.0f);
  proj_gemm_kernel<<<dim3(8, 16), 128, 0, stream>>>(posh, Wkh, bk, pkh, 1.0f);
  proj_gemm_kernel<<<dim3(8, 16), 128, 0, stream>>>(posh, Wqh, bq, pqh, iscale);

  // 4) position-score matrices: c2p = Q . pos_k^T, p2c = K . pos_q^T
  posscore_gemm_kernel<<<dim3(16, 8, 64), 128, 0, stream>>>(Qh, pkh, c2pw);
  posscore_gemm_kernel<<<dim3(16, 8, 64), 128, 0, stream>>>(Kh, pqh, p2cw);

  // 5) fused attention (QK^T + gathers + mask + softmax + PV)
  attn_kernel<<<dim3(16, 16, 4), 128, 0, stream>>>(Qh, Kh, Vh, c2pw, p2cw, am, idxt, out);
}
